// SVGLSTMEncoder_2662879724018
// MI455X (gfx1250) — matrix operand order repaired and checked by reading, not compiled, not run on hardware
//
#include <hip/hip_runtime.h>
#include <hip/hip_bf16.h>

// ---------------------------------------------------------------------------
// LayerNorm-LSTM (L=4, T=51, B=128, H=1024) for gfx1250 (MI455X).
// GEMMs: v_wmma_f32_16x16x32_bf16 with double-buffered async global->LDS
// staging (GLOBAL_LOAD_ASYNC_TO_LDS_B128, ASYNCcnt-tracked).
// ---------------------------------------------------------------------------

typedef __bf16 bf16_t;
typedef __attribute__((ext_vector_type(16))) __bf16 v16bf;
typedef __attribute__((ext_vector_type(8)))  __bf16 v8bf;
typedef __attribute__((ext_vector_type(8)))  float  v8f;
typedef __attribute__((address_space(3))) bf16_t lds_bf16_t;

#define LAYERS  4
#define T_STEPS 51
#define BATCH   128
#define HID     1024
#define G4H     (4 * HID)      // 4096
#define LN_EPS  1e-5f

// Async DMA: global (16B) -> LDS, no VGPR round trip, tracked by ASYNCcnt.
__device__ __forceinline__ void async_ld16(const bf16_t* g, bf16_t* l) {
  const unsigned lds_off = (unsigned)(unsigned long long)(lds_bf16_t*)l;
  const unsigned long long ga = (unsigned long long)(const void*)g;
  asm volatile("global_load_async_to_lds_b128 %0, %1, off"
               :: "v"(lds_off), "v"(ga) : "memory");
}

// ---------------------------------------------------------------------------
// Tiled WMMA GEMM:  C[M,N] = A[M,K] * W[N,K]^T  (gates = x @ w^T)
// A, W bf16 row-major; f32 accumulation; output f32 or bf16.
// WG = 256 threads = 8 wave32 arranged 2(M) x 4(N); wave tile 32M x 32N
// (4 accumulators -> 8 WMMAs per 64-wide K slice per wave).
// WG tile 64M x 128N. K staged via async-to-LDS double buffering.
// Requires M % 64 == 0, N % 128 == 0, K % 64 == 0
// (true here: M in {6528,128}, N = 4096, K = 1024).
// ---------------------------------------------------------------------------
template <bool OUT_BF16>
__global__ __launch_bounds__(256) void gemm_xwt_wmma(
    const bf16_t* __restrict__ A, const bf16_t* __restrict__ W,
    void* __restrict__ Cout, int M, int N, int K) {
  __shared__ alignas(16) bf16_t Asm[2][64][72];   // +8 pad, stride 144B
  __shared__ alignas(16) bf16_t Bsm[2][128][72];

  const int tid  = threadIdx.x;
  const int lane = tid & 31;   // wave32
  const int wave = tid >> 5;   // 0..7
  const int wm   = wave >> 2;  // 0..1  (M)
  const int wn   = wave & 3;   // 0..3  (N)

  const int m0 = blockIdx.y * 64;
  const int n0 = blockIdx.x * 128;

  v8f acc[2][2] = {};

  // 6 async 16B transfers per thread per 64-wide K slice (2 for A, 4 for W).
  auto stage = [&](int buf, int kb) {
#pragma unroll
    for (int i = 0; i < 2; ++i) {
      const int eid = tid + 256 * i;
      const int r = eid >> 3, c8 = (eid & 7) * 8;
      async_ld16(&A[(size_t)(m0 + r) * K + kb + c8], &Asm[buf][r][c8]);
    }
#pragma unroll
    for (int i = 0; i < 4; ++i) {
      const int eid = tid + 256 * i;
      const int r = eid >> 3, c8 = (eid & 7) * 8;
      async_ld16(&W[(size_t)(n0 + r) * K + kb + c8], &Bsm[buf][r][c8]);
    }
  };

  stage(0, 0);
  for (int kb = 0; kb < K; kb += 64) {
    const int cur = (kb >> 6) & 1;
    const bool more = (kb + 64) < K;
    if (more) {
      stage(cur ^ 1, kb + 64);  // prefetch next slice into other buffer
      asm volatile("s_wait_asynccnt 0x6" ::: "memory");  // cur slice landed
    } else {
      asm volatile("s_wait_asynccnt 0x0" ::: "memory");
    }
    __syncthreads();

#pragma unroll
    for (int kk = 0; kk < 64; kk += 32) {
      const int ksel = lane >> 4;
      v16bf av[2], bv[2];
      // A operand (16x32 bf16, ISA 7.12.2): lane&15 = M row; lane halves take
      // interleaved 8-elem K chunks: lo lanes K 0-7/16-23, hi 8-15/24-31.
#pragma unroll
      for (int mi = 0; mi < 2; ++mi) {
        const bf16_t* ar = &Asm[cur][wm * 32 + mi * 16 + (lane & 15)][0];
        v8bf lo = *reinterpret_cast<const v8bf*>(&ar[kk + ksel * 8]);
        v8bf hi = *reinterpret_cast<const v8bf*>(&ar[kk + 16 + ksel * 8]);
#pragma unroll
        for (int i = 0; i < 8; ++i) { av[mi][i] = lo[i]; av[mi][8 + i] = hi[i]; }
      }
      // B operand (32x16 bf16): lane&15 = N column (= row of W); lane halves
      // take interleaved 8-elem K chunks: lo lanes K 0-7/16-23, hi 8-15/24-31.
#pragma unroll
      for (int ni = 0; ni < 2; ++ni) {
        const bf16_t* br = &Bsm[cur][wn * 32 + ni * 16 + (lane & 15)][0];
        v8bf lo = *reinterpret_cast<const v8bf*>(&br[kk + ksel * 8]);
        v8bf hi = *reinterpret_cast<const v8bf*>(&br[kk + 16 + ksel * 8]);
#pragma unroll
        for (int i = 0; i < 8; ++i) { bv[ni][i] = lo[i]; bv[ni][8 + i] = hi[i]; }
      }
#pragma unroll
      for (int mi = 0; mi < 2; ++mi)
#pragma unroll
        for (int ni = 0; ni < 2; ++ni)
          acc[mi][ni] = __builtin_amdgcn_wmma_f32_16x16x32_bf16(
              false, av[mi], false, bv[ni], (short)0, acc[mi][ni], false, false);
    }
    __syncthreads();  // buf[cur] fully consumed before it is restaged
  }

  // C/D layout: VGPR r -> M = r (lanes 0-15) or 8+r (lanes 16-31), N = lane&15
#pragma unroll
  for (int mi = 0; mi < 2; ++mi) {
#pragma unroll
    for (int ni = 0; ni < 2; ++ni) {
      const int col   = n0 + wn * 32 + ni * 16 + (lane & 15);
      const int rbase = m0 + wm * 32 + mi * 16 + ((lane >> 4) ? 8 : 0);
      if (OUT_BF16) {
        bf16_t* C = reinterpret_cast<bf16_t*>(Cout);
#pragma unroll
        for (int r = 0; r < 8; ++r)
          C[(size_t)(rbase + r) * N + col] = (bf16_t)acc[mi][ni][r];
      } else {
        float* C = reinterpret_cast<float*>(Cout);
#pragma unroll
        for (int r = 0; r < 8; ++r)
          C[(size_t)(rbase + r) * N + col] = acc[mi][ni][r];
      }
    }
  }
}

// ---------------------------------------------------------------------------
__device__ __forceinline__ float block_reduce_sum(float v, float* sbuf) {
  const int tid = threadIdx.x;
  sbuf[tid] = v;
  __syncthreads();
#pragma unroll
  for (int s = 128; s > 0; s >>= 1) {
    if (tid < s) sbuf[tid] += sbuf[tid + s];
    __syncthreads();
  }
  const float r = sbuf[0];
  __syncthreads();
  return r;
}

__device__ __forceinline__ float sigm(float x) {
  return 1.f / (1.f + __expf(-x));
}

// In-place LayerNorm over 4096-wide bf16 rows (one block per row).
__global__ __launch_bounds__(256) void ln_rows_bf16(
    bf16_t* __restrict__ X, const float* __restrict__ g,
    const float* __restrict__ bta) {
  __shared__ float sbuf[256];
  const size_t row = (size_t)blockIdx.x * G4H;
  const int tid = threadIdx.x;
  float v[16], s = 0.f, sq = 0.f;
#pragma unroll
  for (int q = 0; q < 16; ++q) {
    v[q] = (float)X[row + tid + 256 * q];
    s += v[q];
    sq += v[q] * v[q];
  }
  const float mean = block_reduce_sum(s, sbuf) * (1.f / G4H);
  const float var  = block_reduce_sum(sq, sbuf) * (1.f / G4H) - mean * mean;
  const float rs   = rsqrtf(var + LN_EPS);
#pragma unroll
  for (int q = 0; q < 16; ++q) {
    const int j = tid + 256 * q;
    X[row + j] = (bf16_t)((v[q] - mean) * rs * g[j] + bta[j]);
  }
}

// Fused LSTM step: LN(hgates) + ig -> gates -> cell update -> LN(c) -> h.
// One block per batch row.
__global__ __launch_bounds__(256) void lstm_step(
    const float* __restrict__ hg,     // [B,4H] raw h-side gates
    const bf16_t* __restrict__ ig_t,  // [B,4H] LN'd input gates (this t)
    const float* __restrict__ g_hh, const float* __restrict__ b_hh,  // [4H]
    const float* __restrict__ g_c,  const float* __restrict__ b_c,   // [H]
    float* __restrict__ h_cur, float* __restrict__ c_cur,            // [B,H]
    bf16_t* __restrict__ h_bf,    // [B,H] bf16 h for next-step GEMM
    bf16_t* __restrict__ x_next,  // [B,H] slot t of next layer's input
    float* __restrict__ x_out) {  // [B,H] slot t of d_out 'output'
  __shared__ float sbuf[256];
  const int b = blockIdx.x, tid = threadIdx.x;
  const float* row = hg + (size_t)b * G4H;

  float s = 0.f, sq = 0.f;
#pragma unroll
  for (int q = 0; q < 16; ++q) {
    const float v = row[tid + 256 * q];
    s += v;
    sq += v * v;
  }
  const float mean = block_reduce_sum(s, sbuf) * (1.f / G4H);
  const float var  = block_reduce_sum(sq, sbuf) * (1.f / G4H) - mean * mean;
  const float rs   = rsqrtf(var + LN_EPS);

  float cn[4], og[4], s2 = 0.f, sq2 = 0.f;
#pragma unroll
  for (int q = 0; q < 4; ++q) {
    const int j = tid + 256 * q;
    float gate[4];
#pragma unroll
    for (int k = 0; k < 4; ++k) {
      const int idx = j + k * HID;
      const float hv = (row[idx] - mean) * rs * g_hh[idx] + b_hh[idx];
      gate[k] = hv + (float)ig_t[(size_t)b * G4H + idx];
    }
    const float c_new =
        sigm(gate[1]) * c_cur[(size_t)b * HID + j] + sigm(gate[0]) * tanhf(gate[2]);
    cn[q] = c_new;
    og[q] = gate[3];
    s2 += c_new;
    sq2 += c_new * c_new;
  }
  const float mean2 = block_reduce_sum(s2, sbuf) * (1.f / HID);
  const float var2  = block_reduce_sum(sq2, sbuf) * (1.f / HID) - mean2 * mean2;
  const float rs2   = rsqrtf(var2 + LN_EPS);
#pragma unroll
  for (int q = 0; q < 4; ++q) {
    const int j = tid + 256 * q;
    const float chat = (cn[q] - mean2) * rs2 * g_c[j] + b_c[j];
    const float h = sigm(og[q]) * tanhf(chat);
    const size_t o = (size_t)b * HID + j;
    c_cur[o] = cn[q];
    h_cur[o] = h;
    h_bf[o] = (bf16_t)h;
    x_next[o] = (bf16_t)h;
    x_out[o] = h;
  }
}

// ---------------------------------------------------------------------------
__global__ void f2b(const float* __restrict__ s, bf16_t* __restrict__ d, int n) {
  const int i = blockIdx.x * blockDim.x + threadIdx.x;
  if (i < n) d[i] = (bf16_t)s[i];
}

__global__ void init_hc(const float* __restrict__ h0, const float* __restrict__ c0,
                        float* __restrict__ h, float* __restrict__ c,
                        bf16_t* __restrict__ hb, int n) {
  const int i = blockIdx.x * blockDim.x + threadIdx.x;
  if (i < n) {
    const float hv = h0[i];
    h[i] = hv;
    hb[i] = (bf16_t)hv;
    c[i] = c0[i];
  }
}

__global__ void copy_f(const float* __restrict__ s, float* __restrict__ d, int n) {
  const int i = blockIdx.x * blockDim.x + threadIdx.x;
  if (i < n) d[i] = s[i];
}

// ---------------------------------------------------------------------------
extern "C" void kernel_launch(void* const* d_in, const int* in_sizes, int n_in,
                              void* d_out, int out_size, void* d_ws,
                              size_t ws_size, hipStream_t stream) {
  const float* inpt   = (const float*)d_in[0];   // [T,B,H]
  const float* hidden = (const float*)d_in[1];   // [L,B,H]
  const float* cell   = (const float*)d_in[2];   // [L,B,H]
  const float* w_ih   = (const float*)d_in[3];   // [L,4H,H]
  const float* w_hh   = (const float*)d_in[4];   // [L,4H,H]
  const float* g_ih   = (const float*)d_in[5];   // [L,4H]
  const float* b_ih   = (const float*)d_in[6];
  const float* g_hh   = (const float*)d_in[7];
  const float* b_hh   = (const float*)d_in[8];
  const float* g_c    = (const float*)d_in[9];   // [L,H]
  const float* b_c    = (const float*)d_in[10];

  float* out_x = (float*)d_out;                                  // [T,B,H]
  float* out_h = out_x + (size_t)T_STEPS * BATCH * HID;          // [L,B,H]
  float* out_c = out_h + (size_t)LAYERS * BATCH * HID;           // [L,B,H]

  // Workspace carve-out (~131 MB total).
  char* wsb = (char*)d_ws;
  size_t off = 0;
  auto carve = [&](size_t bytes) -> void* {
    void* p = wsb + off;
    off += (bytes + 255) & ~(size_t)255;
    return p;
  };
  bf16_t* wih_bf = (bf16_t*)carve((size_t)LAYERS * G4H * HID * 2);
  bf16_t* whh_bf = (bf16_t*)carve((size_t)LAYERS * G4H * HID * 2);
  bf16_t* x_bf   = (bf16_t*)carve((size_t)T_STEPS * BATCH * HID * 2);
  bf16_t* ig_bf  = (bf16_t*)carve((size_t)T_STEPS * BATCH * G4H * 2);
  float*  hgates = (float*)carve((size_t)BATCH * G4H * 4);
  float*  h_cur  = (float*)carve((size_t)BATCH * HID * 4);
  float*  c_cur  = (float*)carve((size_t)BATCH * HID * 4);
  bf16_t* h_bf   = (bf16_t*)carve((size_t)BATCH * HID * 2);
  if (off > ws_size) return;  // workspace too small; nothing safe to do

  const int nw = LAYERS * G4H * HID;
  f2b<<<(nw + 255) / 256, 256, 0, stream>>>(w_ih, wih_bf, nw);
  f2b<<<(nw + 255) / 256, 256, 0, stream>>>(w_hh, whh_bf, nw);
  const int nx = T_STEPS * BATCH * HID;
  f2b<<<(nx + 255) / 256, 256, 0, stream>>>(inpt, x_bf, nx);

  const int nh = BATCH * HID;
  for (int l = 0; l < LAYERS; ++l) {
    init_hc<<<(nh + 255) / 256, 256, 0, stream>>>(
        hidden + (size_t)l * nh, cell + (size_t)l * nh, h_cur, c_cur, h_bf, nh);

    // Input-side gates for all timesteps: [T*B,4H] = x @ w_ih[l]^T, then LN.
    dim3 gi(G4H / 128, (T_STEPS * BATCH) / 64);
    gemm_xwt_wmma<true><<<gi, 256, 0, stream>>>(
        x_bf, wih_bf + (size_t)l * G4H * HID, ig_bf, T_STEPS * BATCH, G4H, HID);
    ln_rows_bf16<<<T_STEPS * BATCH, 256, 0, stream>>>(
        ig_bf, g_ih + (size_t)l * G4H, b_ih + (size_t)l * G4H);

    // Recurrent chain (serialized on the stream; graph replay amortizes it).
    dim3 gh(G4H / 128, BATCH / 64);
    for (int t = 0; t < T_STEPS; ++t) {
      gemm_xwt_wmma<false><<<gh, 256, 0, stream>>>(
          h_bf, whh_bf + (size_t)l * G4H * HID, hgates, BATCH, G4H, HID);
      lstm_step<<<BATCH, 256, 0, stream>>>(
          hgates, ig_bf + (size_t)t * BATCH * G4H,
          g_hh + (size_t)l * G4H, b_hh + (size_t)l * G4H,
          g_c + (size_t)l * HID, b_c + (size_t)l * HID,
          h_cur, c_cur, h_bf,
          x_bf + (size_t)t * BATCH * HID,      // becomes next layer's input
          out_x + (size_t)t * BATCH * HID);    // final layer's values survive
    }
    copy_f<<<(nh + 255) / 256, 256, 0, stream>>>(h_cur, out_h + (size_t)l * nh, nh);
    copy_f<<<(nh + 255) / 256, 256, 0, stream>>>(c_cur, out_c + (size_t)l * nh, nh);
  }
}
